// RNNLanguage_37263136260392
// MI455X (gfx1250) — compile-verified, matrix operand order checked by reading, not run on hardware
//
#include <hip/hip_runtime.h>
#include <hip/hip_bf16.h>
#include <stdint.h>

// ---------------- problem constants ----------------
constexpr int B_ = 32;
constexpr int T_ = 512;
constexpr int H_ = 1024;
constexpr int V_ = 50257;
constexpr int O_ = 50257;
constexpr int NT_O = (O_ + 15) / 16;   // 3142 N-tiles over the vocab dim
constexpr int NB1  = 128;              // K1 workgroups (8 waves each -> 1024 waves)

// Large finite sentinel: behaves like -inf in logsumexp merges (its exp weight
// underflows to exactly 0 against any real value) but never produces NaN, so
// all merge code is branch-free (no v_cmpx / exec-mask churn in the hot loop).
constexpr float NEG_BIG = -1.0e30f;

// ---------------- CDNA5 WMMA types ----------------
typedef __attribute__((ext_vector_type(16))) __bf16 v16bf;
typedef __attribute__((ext_vector_type(8)))  float  v8f;

static __device__ __forceinline__ v8f wmma_bf16(v16bf a, v16bf b, v8f c) {
  // D = A(16x32 bf16) x B(32x16 bf16) + C(16x16 f32)
  return __builtin_amdgcn_wmma_f32_16x16x32_bf16(false, a, false, b, (short)0, c,
                                                 false, false);
}

// Load 16 bf16 elements for a WMMA A/B fragment: two contiguous 16B segments
// at K-offsets [off, off+8) and [off+16, off+24) within a row.
static __device__ __forceinline__ v16bf load_frag16(const uint16_t* __restrict__ row,
                                                    int off) {
  union { uint4 q[2]; v16bf v; } f;
  f.q[0] = *reinterpret_cast<const uint4*>(row + off);
  f.q[1] = *reinterpret_cast<const uint4*>(row + off + 16);
  return f.v;
}

// ---------------- small numeric helpers ----------------
static __device__ __forceinline__ uint16_t f2bf(float f) {
  union { float f; uint32_t u; } c{f};
  uint32_t x = c.u;
  uint32_t r = x + 0x7FFFu + ((x >> 16) & 1u);   // round-to-nearest-even
  return (uint16_t)(r >> 16);
}
static __device__ __forceinline__ float bf2f(uint16_t h) {
  union { uint32_t u; float f; } c;
  c.u = ((uint32_t)h) << 16;
  return c.f;
}
// Branch-free merge of two (max, sumexp) streaming-logsumexp pairs.
// Safe because all m values are finite (NEG_BIG sentinel, never -inf).
static __device__ __forceinline__ void lse_combine(float& m, float& s, float m2, float s2) {
  float nm = fmaxf(m, m2);
  s = s * __expf(m - nm) + s2 * __expf(m2 - nm);
  m = nm;
}
static __device__ __forceinline__ float wave_sum(float v) {
#pragma unroll
  for (int off = 16; off; off >>= 1) v += __shfl_xor(v, off, 32);
  return v;
}
static __device__ __forceinline__ float wave_max(float v) {
#pragma unroll
  for (int off = 16; off; off >>= 1) v = fmaxf(v, __shfl_xor(v, off, 32));
  return v;
}

// =====================================================================
// Preprocessing kernels (run once per kernel_launch)
// =====================================================================

// init: copy input state into evolving f32/bf16 state buffers, zero loss
__global__ void rnn_init(const float* __restrict__ state, float* __restrict__ st,
                         uint16_t* __restrict__ stb, float* __restrict__ loss) {
  int i = blockIdx.x * blockDim.x + threadIdx.x;
  if (i < B_ * H_) { float v = state[i]; st[i] = v; stb[i] = f2bf(v); }
  if (i == 0) loss[0] = 0.f;
}

// tiled transpose + fp32->bf16:  out_w[H][O] -> out_wT[O][H]
__global__ void rnn_transpose_w(const float* __restrict__ w, uint16_t* __restrict__ wT) {
  __shared__ float tile[32][33];
  const int tO = blockIdx.x, tH = blockIdx.y;
  const int lx = threadIdx.x, ly = threadIdx.y;     // (32, 8)
#pragma unroll
  for (int i = 0; i < 4; ++i) {
    int h = tH * 32 + ly + 8 * i;
    int o = tO * 32 + lx;
    tile[ly + 8 * i][lx] = (o < O_) ? w[(size_t)h * O_ + o] : 0.f;
  }
  __syncthreads();
#pragma unroll
  for (int i = 0; i < 4; ++i) {
    int o = tO * 32 + ly + 8 * i;
    int h = tH * 32 + lx;
    if (o < O_) wT[(size_t)o * H_ + h] = f2bf(tile[lx][ly + 8 * i]);
  }
}

// norms[r] = || hh[r,:] ||  (one wave per row)
__global__ void rnn_rownorm(const float* __restrict__ hh, float* __restrict__ norms) {
  int lane = threadIdx.x & 31, wave = threadIdx.x >> 5;
  int gw = blockIdx.x * (blockDim.x >> 5) + wave;
  if (gw >= H_) return;
  float acc = 0.f;
  for (int h = lane; h < H_; h += 32) { float v = hh[(size_t)gw * H_ + h]; acc += v * v; }
  acc = wave_sum(acc);
  if (lane == 0) norms[gw] = sqrtf(acc);
}

// hhT[j][i] = hh[i][j] / (norms[j]*H)   (broadcast over LAST axis, as in ref)
__global__ void rnn_scale_hh(const float* __restrict__ hh, const float* __restrict__ norms,
                             uint16_t* __restrict__ hhT) {
  __shared__ float tile[32][33];
  const int ti = blockIdx.x, tj = blockIdx.y;
  const int lx = threadIdx.x, ly = threadIdx.y;     // (32, 8)
#pragma unroll
  for (int k = 0; k < 4; ++k) {
    int i = ti * 32 + ly + 8 * k;
    int j = tj * 32 + lx;
    tile[ly + 8 * k][lx] = hh[(size_t)i * H_ + j];
  }
  __syncthreads();
#pragma unroll
  for (int k = 0; k < 4; ++k) {
    int j = tj * 32 + ly + 8 * k;
    int i = ti * 32 + lx;
    float scale = 1.0f / (norms[j] * (float)H_);
    hhT[(size_t)j * H_ + i] = f2bf(tile[lx][ly + 8 * k] * scale);
  }
}

// ihrow[i] = sum_j ih[i][j]   (one wave per row)
__global__ void rnn_ihrow(const float* __restrict__ ih, float* __restrict__ ihrow) {
  int lane = threadIdx.x & 31, wave = threadIdx.x >> 5;
  int gw = blockIdx.x * (blockDim.x >> 5) + wave;
  if (gw >= H_) return;
  float acc = 0.f;
  for (int j = lane; j < H_; j += 32) acc += ih[(size_t)gw * H_ + j];
  acc = wave_sum(acc);
  if (lane == 0) ihrow[gw] = acc;
}

__global__ void rnn_sumbias(const float* __restrict__ bias, float* __restrict__ sb) {
  __shared__ float sm[256];
  float a = 0.f;
  for (int i = threadIdx.x; i < H_; i += 256) a += bias[i];
  sm[threadIdx.x] = a;
  __syncthreads();
  for (int s = 128; s; s >>= 1) {
    if (threadIdx.x < s) sm[threadIdx.x] += sm[threadIdx.x + s];
    __syncthreads();
  }
  if (threadIdx.x == 0) sb[0] = sm[0];
}

// esum[b*T+t] = tanh(embedding[tok,:]) . ihrow + sumBias   (one wave per token)
__global__ void rnn_esum(const int* __restrict__ tokens, const float* __restrict__ emb,
                         const float* __restrict__ ihrow, const float* __restrict__ sb,
                         float* __restrict__ esum) {
  int lane = threadIdx.x & 31, wave = threadIdx.x >> 5;
  int gw = blockIdx.x * (blockDim.x >> 5) + wave;
  int GW = gridDim.x * (blockDim.x >> 5);
  for (int idx = gw; idx < B_ * T_; idx += GW) {
    int v = tokens[idx];
    float acc = 0.f;
    for (int h = lane; h < H_; h += 32)
      acc += tanhf(emb[(size_t)v * H_ + h]) * ihrow[h];
    acc = wave_sum(acc);
    if (lane == 0) esum[idx] = acc + sb[0];
  }
}

// =====================================================================
// Per-step kernels
// =====================================================================

// K1: logits = st @ out_wT^T + outBias, streamed through WMMA with a per-lane
// online logsumexp over the vocab dimension. Writes per-block (m,s) partials.
__global__ void rnn_logits_lse(const uint16_t* __restrict__ stb,
                               const uint16_t* __restrict__ outwT,
                               const float* __restrict__ outBias,
                               float* __restrict__ partials) {
  const int lane = threadIdx.x & 31;
  const int wave = threadIdx.x >> 5;
  const int gw = blockIdx.x * (blockDim.x >> 5) + wave;
  const int GW = gridDim.x * (blockDim.x >> 5);
  const int half = lane >> 4;
  const int l15 = lane & 15;
  const int klo = half ? 8 : 0;

  // A rows for this lane (st is 32 x 1024 bf16, row-major)
  const uint16_t* arow0 = stb + (size_t)l15 * H_;        // M-tile 0: b = l15
  const uint16_t* arow1 = arow0 + (size_t)16 * H_;       // M-tile 1: b = 16 + l15

  float m[16], s[16];
#pragma unroll
  for (int i = 0; i < 16; ++i) { m[i] = NEG_BIG; s[i] = 0.f; }

  for (int nt = gw; nt < NT_O; nt += GW) {
    const int ncol = nt * 16 + l15;
    const bool valid = (ncol < O_);
    const uint16_t* brow = outwT + (size_t)(valid ? ncol : 0) * H_;

    // prefetch next tile's B row into the cache hierarchy
    int nnt = nt + GW;
    if (nnt < NT_O) {
      int nn = nnt * 16 + l15;
      if (nn < O_) __builtin_prefetch(outwT + (size_t)nn * H_, 0, 1);
    }

    v8f acc0 = {}; v8f acc1 = {};
#pragma unroll 4
    for (int ks = 0; ks < H_ / 32; ++ks) {
      const int kb = ks * 32 + klo;
      v16bf a0 = load_frag16(arow0, kb);
      v16bf a1 = load_frag16(arow1, kb);
      v16bf bb = load_frag16(brow, kb);
      acc0 = wmma_bf16(a0, bb, acc0);
      acc1 = wmma_bf16(a1, bb, acc1);
    }
    float bias = valid ? outBias[ncol] : 0.f;
#pragma unroll
    for (int r = 0; r < 8; ++r) {
      float x0 = valid ? (acc0[r] + bias) : NEG_BIG;   // b = r + half*8
      float x1 = valid ? (acc1[r] + bias) : NEG_BIG;   // b = 16 + r + half*8
      lse_combine(m[r],     s[r],     x0, 1.f);
      lse_combine(m[8 + r], s[8 + r], x1, 1.f);
    }
  }

  // merge across the 16 lanes of each half (xor of bits 0..3 stays in-half)
#pragma unroll
  for (int i = 0; i < 16; ++i) {
#pragma unroll
    for (int off = 1; off < 16; off <<= 1) {
      float m2 = __shfl_xor(m[i], off, 32);
      float s2 = __shfl_xor(s[i], off, 32);
      lse_combine(m[i], s[i], m2, s2);
    }
  }

  __shared__ float sm[8][32], ss[8][32];
  if (l15 == 0) {
#pragma unroll
    for (int r = 0; r < 8; ++r) {
      sm[wave][r + half * 8]      = m[r];     ss[wave][r + half * 8]      = s[r];
      sm[wave][16 + r + half * 8] = m[8 + r]; ss[wave][16 + r + half * 8] = s[8 + r];
    }
  }
  __syncthreads();
  if (threadIdx.x < 32) {
    int b = threadIdx.x;
    float M = sm[0][b], S = ss[0][b];
    for (int w = 1; w < (int)(blockDim.x >> 5); ++w) lse_combine(M, S, sm[w][b], ss[w][b]);
    partials[blockIdx.x * 64 + b * 2]     = M;
    partials[blockIdx.x * 64 + b * 2 + 1] = S;
  }
}

// K2: merge block partials -> logsumexp per row, compute target logit, add to loss
__global__ void rnn_loss(const float* __restrict__ partials,
                         const uint16_t* __restrict__ stb,
                         const uint16_t* __restrict__ outwT,
                         const float* __restrict__ outBias,
                         const int* __restrict__ tokens, int t,
                         float* __restrict__ loss) {
  __shared__ float lse[32], tl[32];
  if (threadIdx.x < 32) {
    int b = threadIdx.x;
    float M = NEG_BIG, S = 0.f;
    for (int blk = 0; blk < NB1; ++blk)
      lse_combine(M, S, partials[blk * 64 + b * 2], partials[blk * 64 + b * 2 + 1]);
    lse[b] = M + __logf(S);
  }
  int wave = threadIdx.x >> 5, lane = threadIdx.x & 31;
  for (int b = wave; b < 32; b += (int)(blockDim.x >> 5)) {
    int tok = tokens[b * T_ + t];
    const uint16_t* wrow = outwT + (size_t)tok * H_;
    const uint16_t* srow = stb + (size_t)b * H_;
    float acc = 0.f;
    for (int h = lane; h < H_; h += 32) acc += bf2f(srow[h]) * bf2f(wrow[h]);
    acc = wave_sum(acc);
    if (lane == 0) tl[b] = acc + outBias[tok];
  }
  __syncthreads();
  if (threadIdx.x == 0) {
    float step = 0.f;
    for (int b = 0; b < 32; ++b) step += lse[b] - tl[b];
    loss[0] += step * (1.0f / (float)B_);
  }
}

// K3a: attn = softmax(st * esum_b) over H, one wave per batch row, bf16 out
__global__ void rnn_attn(const float* __restrict__ st, const float* __restrict__ esum,
                         int t, uint16_t* __restrict__ attnb) {
  int lane = threadIdx.x & 31, b = threadIdx.x >> 5;   // 32 waves = 32 rows
  float e = esum[b * T_ + t];
  float x[32];
  float mx = NEG_BIG;
#pragma unroll
  for (int i = 0; i < 32; ++i) {
    x[i] = st[(size_t)b * H_ + lane + i * 32] * e;
    mx = fmaxf(mx, x[i]);
  }
  mx = wave_max(mx);
  float sum = 0.f;
#pragma unroll
  for (int i = 0; i < 32; ++i) { x[i] = __expf(x[i] - mx); sum += x[i]; }
  sum = wave_sum(sum);
  float inv = 1.0f / sum;
#pragma unroll
  for (int i = 0; i < 32; ++i)
    attnb[(size_t)b * H_ + lane + i * 32] = f2bf(x[i] * inv);
}

// K3b: st = tanh(st + attn @ scaledHH) via WMMA (one wave per 16-col N-tile)
__global__ void rnn_update(float* __restrict__ st, uint16_t* __restrict__ stb,
                           const uint16_t* __restrict__ attnb,
                           const uint16_t* __restrict__ hhT) {
  const int lane = threadIdx.x & 31;
  const int wave = threadIdx.x >> 5;
  const int nt = blockIdx.x * (blockDim.x >> 5) + wave;  // 0..63
  const int half = lane >> 4, l15 = lane & 15;
  const int klo = half ? 8 : 0;

  const uint16_t* arow0 = attnb + (size_t)l15 * H_;
  const uint16_t* arow1 = arow0 + (size_t)16 * H_;
  const uint16_t* brow  = hhT + (size_t)(nt * 16 + l15) * H_;

  v8f acc0 = {}; v8f acc1 = {};
#pragma unroll 4
  for (int ks = 0; ks < H_ / 32; ++ks) {
    const int kb = ks * 32 + klo;
    v16bf a0 = load_frag16(arow0, kb);
    v16bf a1 = load_frag16(arow1, kb);
    v16bf bb = load_frag16(brow, kb);
    acc0 = wmma_bf16(a0, bb, acc0);
    acc1 = wmma_bf16(a1, bb, acc1);
  }
  const int n = nt * 16 + l15;
#pragma unroll
  for (int r = 0; r < 8; ++r) {
    int b0 = r + half * 8;
    int b1 = 16 + r + half * 8;
    float v0 = tanhf(st[(size_t)b0 * H_ + n] + acc0[r]);
    float v1 = tanhf(st[(size_t)b1 * H_ + n] + acc1[r]);
    st[(size_t)b0 * H_ + n] = v0; stb[(size_t)b0 * H_ + n] = f2bf(v0);
    st[(size_t)b1 * H_ + n] = v1; stb[(size_t)b1 * H_ + n] = f2bf(v1);
  }
}

// final: pack (state[32*1024], loss) into d_out
__global__ void rnn_output(const float* __restrict__ st, const float* __restrict__ loss,
                           float* __restrict__ out) {
  int i = blockIdx.x * blockDim.x + threadIdx.x;
  if (i < B_ * H_) out[i] = st[i];
  if (i == 0) out[B_ * H_] = loss[0];
}

// =====================================================================
// host-side launch
// =====================================================================
extern "C" void kernel_launch(void* const* d_in, const int* in_sizes, int n_in,
                              void* d_out, int out_size, void* d_ws, size_t ws_size,
                              hipStream_t stream) {
  (void)in_sizes; (void)n_in; (void)out_size; (void)ws_size;
  const float* state   = (const float*)d_in[0];
  const int*   tokens  = (const int*)  d_in[1];
  const float* emb     = (const float*)d_in[2];
  const float* ih      = (const float*)d_in[3];
  const float* hh      = (const float*)d_in[4];
  const float* bias    = (const float*)d_in[5];
  const float* out_w   = (const float*)d_in[6];
  const float* outBias = (const float*)d_in[7];
  float* out = (float*)d_out;

  // workspace carve-up (all 256B aligned)
  uint8_t* ws = (uint8_t*)d_ws;
  size_t cur = 0;
  auto carve = [&](size_t bytes) { void* p = ws + cur; cur += (bytes + 255) & ~(size_t)255; return p; };
  uint16_t* outwT  = (uint16_t*)carve((size_t)O_ * H_ * 2);  // 103 MB, L2-resident
  uint16_t* hhT    = (uint16_t*)carve((size_t)H_ * H_ * 2);  // 2 MB
  float*    st     = (float*)   carve((size_t)B_ * H_ * 4);
  uint16_t* stb    = (uint16_t*)carve((size_t)B_ * H_ * 2);
  uint16_t* attnb  = (uint16_t*)carve((size_t)B_ * H_ * 2);
  float*    esum   = (float*)   carve((size_t)B_ * T_ * 4);
  float*    ihrow  = (float*)   carve((size_t)H_ * 4);
  float*    norms  = (float*)   carve((size_t)H_ * 4);
  float*    sumb   = (float*)   carve(256);
  float*    part   = (float*)   carve((size_t)NB1 * 64 * 4);
  float*    loss   = (float*)   carve(256);

  // ---- preprocessing ----
  rnn_init<<<(B_ * H_ + 255) / 256, 256, 0, stream>>>(state, st, stb, loss);
  rnn_transpose_w<<<dim3((O_ + 31) / 32, H_ / 32), dim3(32, 8), 0, stream>>>(out_w, outwT);
  rnn_rownorm<<<H_ / 8, 256, 0, stream>>>(hh, norms);
  rnn_scale_hh<<<dim3(H_ / 32, H_ / 32), dim3(32, 8), 0, stream>>>(hh, norms, hhT);
  rnn_ihrow<<<H_ / 8, 256, 0, stream>>>(ih, ihrow);
  rnn_sumbias<<<1, 256, 0, stream>>>(bias, sumb);
  rnn_esum<<<512, 256, 0, stream>>>(tokens, emb, ihrow, sumb, esum);

  // ---- sequential scan over T ----
  for (int t = 0; t < T_; ++t) {
    rnn_logits_lse<<<NB1, 256, 0, stream>>>(stb, outwT, outBias, part);
    rnn_loss<<<1, 256, 0, stream>>>(part, stb, outwT, outBias, tokens, t, loss);
    rnn_attn<<<1, 1024, 0, stream>>>(st, esum, t, attnb);
    rnn_update<<<8, 256, 0, stream>>>(st, stb, attnb, hhT);
  }

  rnn_output<<<(B_ * H_ + 1 + 255) / 256, 256, 0, stream>>>(st, loss, out);
}
